// SelfAttentionLayer2D_70901320122847
// MI455X (gfx1250) — compile-verified, matrix operand order checked by reading, not run on hardware
//
#include <hip/hip_runtime.h>
#include <hip/hip_bf16.h>
#include <math.h>

typedef __attribute__((ext_vector_type(16))) _Float16 v16h;
typedef __attribute__((ext_vector_type(8)))  float    v8f;

#define BATCH 8
#define NTOK  4096          // 64*64
#define CDIM  128
#define ROWS  (BATCH * NTOK) // 32768

// K index inside a 16x32 (f16) A/B fragment: element e (0..15), half h (lane>>4).
__device__ __forceinline__ int frag_k(int e, int h) {
    return ((e & 8) ? 16 : 0) + (h ? 8 : 0) + (e & 7);
}

// A/B fragment from 16 consecutive rows of a row-major f32 matrix (rows = lane&15).
__device__ __forceinline__ v16h frag_rows_f32(const float* __restrict__ p, int ld) {
    int lane = threadIdx.x & 31;
    const float* row = p + (size_t)(lane & 15) * ld;
    int h = (lane >> 4) & 1;
    v16h f;
#pragma unroll
    for (int e = 0; e < 16; ++e) f[e] = (_Float16)row[frag_k(e, h)];
    return f;
}

// Same, from an f16 row-major matrix.
__device__ __forceinline__ v16h frag_rows_f16(const _Float16* __restrict__ p, int ld) {
    int lane = threadIdx.x & 31;
    const _Float16* row = p + (size_t)(lane & 15) * ld;
    int h = (lane >> 4) & 1;
    v16h f;
#pragma unroll
    for (int e = 0; e < 16; ++e) f[e] = row[frag_k(e, h)];
    return f;
}

// B fragment from 16 consecutive columns of a row-major KxN f32 matrix (weights).
__device__ __forceinline__ v16h frag_cols_f32(const float* __restrict__ p, int ld) {
    int lane = threadIdx.x & 31;
    const float* col = p + (lane & 15);
    int h = (lane >> 4) & 1;
    v16h f;
#pragma unroll
    for (int e = 0; e < 16; ++e) f[e] = (_Float16)col[(size_t)frag_k(e, h) * ld];
    return f;
}

// ---------------------------------------------------------------------------
// Projection: Y = f16( X[ROWS,128] @ W[128,128] + b ).
// MODE 0: Y row-major [ROWS,128].  MODE 1: Y transposed per batch [B][128][4096].
// ---------------------------------------------------------------------------
template <int MODE>
__global__ __launch_bounds__(128) void proj_x_kernel(
    const float* __restrict__ X, const float* __restrict__ W,
    const float* __restrict__ bias, _Float16* __restrict__ Y) {
    int wave = threadIdx.x >> 5;
    int lane = threadIdx.x & 31;
    int row0 = blockIdx.x * 16;
    int col0 = blockIdx.y * 64 + wave * 16;

    v8f acc = {};
#pragma unroll
    for (int k0 = 0; k0 < CDIM; k0 += 32) {
        v16h a = frag_rows_f32(X + (size_t)row0 * CDIM + k0, CDIM);
        v16h b = frag_cols_f32(W + (size_t)k0 * CDIM + col0, CDIM);
        acc = __builtin_amdgcn_wmma_f32_16x16x32_f16(false, a, false, b,
                                                     (short)0, acc, false, false);
    }
    int col = col0 + (lane & 15);
    float bv = bias[col];
    int h = (lane >> 4) & 1;
#pragma unroll
    for (int v = 0; v < 8; ++v) {
        int row = row0 + v + 8 * h;
        float val = acc[v] + bv;
        if (MODE == 0) {
            Y[(size_t)row * CDIM + col] = (_Float16)val;
        } else { // transposed per batch: Y[b][col][n]
            int b_ = row >> 12, n = row & (NTOK - 1);
            Y[((size_t)b_ * CDIM + col) * NTOK + n] = (_Float16)val;
        }
    }
}

// ---------------------------------------------------------------------------
// Pass 1: per (b,n) compute softadj[n] = rowmax + log(rowsum) of fx[n]·gx[m]
// over m.  Per-LANE online stats over that lane's 16-strided column subset;
// cross-lane shuffle merge happens ONCE at the end (keeps the 256-iteration
// inner loop at 4 WMMA + ~6 VALU per row instead of two ds_bpermute trees).
// ---------------------------------------------------------------------------
__global__ __launch_bounds__(256) void row_stats_kernel(
    const _Float16* __restrict__ fx, const _Float16* __restrict__ gx,
    float* __restrict__ softadj) {
    int wave = threadIdx.x >> 5;
    int lane = threadIdx.x & 31;
    int t = blockIdx.x * 8 + wave;
    int b = t >> 8;
    int n0 = (t & 255) * 16;
    const _Float16* fxb = fx + (size_t)b * NTOK * CDIM;
    const _Float16* gxb = gx + (size_t)b * NTOK * CDIM;

    v16h aF[4];
#pragma unroll
    for (int q = 0; q < 4; ++q)
        aF[q] = frag_rows_f16(fxb + (size_t)n0 * CDIM + q * 32, CDIM);

    float lm[8], ls[8];
#pragma unroll
    for (int v = 0; v < 8; ++v) { lm[v] = -3.402823466e38f; ls[v] = 0.f; }

    for (int m0 = 0; m0 < NTOK; m0 += 16) {
        v8f s = {};
#pragma unroll
        for (int q = 0; q < 4; ++q) {
            v16h bB = frag_rows_f16(gxb + (size_t)m0 * CDIM + q * 32, CDIM);
            s = __builtin_amdgcn_wmma_f32_16x16x32_f16(false, aF[q], false, bB,
                                                       (short)0, s, false, false);
        }
#pragma unroll
        for (int v = 0; v < 8; ++v) {           // branchless per-lane online update
            float x = s[v];
            float nm = fmaxf(lm[v], x);
            ls[v] = ls[v] * __expf(lm[v] - nm) + __expf(x - nm);
            lm[v] = nm;
        }
    }
    // merge across the 16 lanes that share each row, then store adj = max+log(sum)
    int h = lane >> 4;
#pragma unroll
    for (int v = 0; v < 8; ++v) {
        float gm = lm[v];
#pragma unroll
        for (int d = 1; d < 16; d <<= 1) gm = fmaxf(gm, __shfl_xor(gm, d, 32));
        float sc = ls[v] * __expf(lm[v] - gm);
#pragma unroll
        for (int d = 1; d < 16; d <<= 1) sc += __shfl_xor(sc, d, 32);
        if ((lane & 15) == 0)
            softadj[(size_t)b * NTOK + n0 + v + 8 * h] = gm + __logf(sc);
    }
}

// ---------------------------------------------------------------------------
// Pass 2: O[b, m-tile, 0:128] = sum_n P^T[m,n] * hx[n,c]; o stored [B][C][N] f16.
// All 8 waves of a block share one batch, so the hx K-chunk (128x32 f16, 8 KB)
// is staged once per block into LDS via global_load_async_to_lds_b128,
// overlapped with the S/P computation, fenced by s_wait_asynccnt + barrier.
// P tiles round-trip per-wave LDS to become A-fragments of P^T.
// ---------------------------------------------------------------------------
__global__ __launch_bounds__(256) void attn_out_kernel(
    const _Float16* __restrict__ fx, const _Float16* __restrict__ gx,
    const _Float16* __restrict__ hxT, const float* __restrict__ softadj,
    _Float16* __restrict__ o) {
    __shared__ _Float16 ldsP[8][32][16]; // per-wave P tile (n_local x m_local)
    __shared__ _Float16 hxS[CDIM][32];   // block-shared hx chunk: [c][n_local]

    int wave = threadIdx.x >> 5;
    int lane = threadIdx.x & 31;
    int t = blockIdx.x * 8 + wave;
    int b = t >> 8;                      // same for all waves in the block
    int m0 = (t & 255) * 16;
    int h = (lane >> 4) & 1;
    int ml = lane & 15;

    const _Float16* fxb  = fx  + (size_t)b * NTOK * CDIM;
    const _Float16* gxb  = gx  + (size_t)b * NTOK * CDIM;
    const _Float16* hxTb = hxT + (size_t)b * CDIM * NTOK;
    const float* adjb = softadj + (size_t)b * NTOK;

    v16h bG[4];
#pragma unroll
    for (int q = 0; q < 4; ++q)
        bG[q] = frag_rows_f16(gxb + (size_t)m0 * CDIM + q * 32, CDIM);

    v8f acc[8];
#pragma unroll
    for (int ct = 0; ct < 8; ++ct) acc[ct] = (v8f){};

    for (int n0 = 0; n0 < NTOK; n0 += 32) {
        // ---- issue async copy of hxT[:, n0:n0+32] into LDS (8 KB, 512 x 16B) ----
#pragma unroll
        for (int rep = 0; rep < 2; ++rep) {
            int q = (int)threadIdx.x + rep * 256;
            int c = q >> 2, seg = q & 3;
            const _Float16* g = hxTb + (size_t)c * NTOK + n0 + seg * 8;
            unsigned lds_off = (unsigned)(size_t)(void*)&hxS[c][seg * 8];
            asm volatile("global_load_async_to_lds_b128 %0, %1, off"
                         :: "v"(lds_off), "v"(g) : "memory");
        }

        // ---- compute P tile for n0..n0+31 into per-wave LDS (overlaps async) ----
#pragma unroll
        for (int sub = 0; sub < 2; ++sub) {
            int n = n0 + sub * 16;
            v8f s = {};
#pragma unroll
            for (int q = 0; q < 4; ++q) {
                v16h aF = frag_rows_f16(fxb + (size_t)n * CDIM + q * 32, CDIM);
                s = __builtin_amdgcn_wmma_f32_16x16x32_f16(false, aF, false, bG[q],
                                                           (short)0, s, false, false);
            }
#pragma unroll
            for (int v = 0; v < 8; ++v) {
                int r = n + v + 8 * h;
                float p = __expf(s[v] - adjb[r]);   // = exp(s-max)/sum
                ldsP[wave][sub * 16 + v + 8 * h][ml] = (_Float16)p;
            }
        }
        asm volatile("" ::: "memory"); // DS in-order per wave covers ldsP RAW

        // A fragment of P^T (16 m-rows x 32 n-cols) from per-wave LDS
        v16h aP;
#pragma unroll
        for (int e = 0; e < 16; ++e) aP[e] = ldsP[wave][frag_k(e, h)][ml];

        // ---- fence the async hx chunk, then 8 PV WMMAs from LDS ----
        asm volatile("s_wait_asynccnt 0" ::: "memory");
        __syncthreads();
#pragma unroll
        for (int ct = 0; ct < 8; ++ct) {
            v16h bH;
#pragma unroll
            for (int e = 0; e < 16; ++e) bH[e] = hxS[ct * 16 + ml][frag_k(e, h)];
            acc[ct] = __builtin_amdgcn_wmma_f32_16x16x32_f16(false, aP, false, bH,
                                                             (short)0, acc[ct], false, false);
        }
        __syncthreads();              // protect hxS against next chunk's writes
    }

    // store O tile: o[b][c][m], c = ct*16 + ml, m = m0 + v + 8h
#pragma unroll
    for (int ct = 0; ct < 8; ++ct) {
        int c = ct * 16 + ml;
#pragma unroll
        for (int v = 0; v < 8; ++v) {
            int m = m0 + v + 8 * h;
            o[((size_t)b * CDIM + c) * NTOK + m] = (_Float16)acc[ct][v];
        }
    }
}

// ---------------------------------------------------------------------------
// Final projection: treat o's flat [B][C*N] buffer as [ROWS,128] row-major
// (exactly the reference's reshape semantics), GEMM with Wv, f32 out + bias.
// ---------------------------------------------------------------------------
__global__ __launch_bounds__(128) void proj_final_kernel(
    const _Float16* __restrict__ X, const float* __restrict__ W,
    const float* __restrict__ bias, float* __restrict__ Y) {
    int wave = threadIdx.x >> 5;
    int lane = threadIdx.x & 31;
    int row0 = blockIdx.x * 16;
    int col0 = blockIdx.y * 64 + wave * 16;

    v8f acc = {};
#pragma unroll
    for (int k0 = 0; k0 < CDIM; k0 += 32) {
        v16h a = frag_rows_f16(X + (size_t)row0 * CDIM + k0, CDIM);
        v16h b = frag_cols_f32(W + (size_t)k0 * CDIM + col0, CDIM);
        acc = __builtin_amdgcn_wmma_f32_16x16x32_f16(false, a, false, b,
                                                     (short)0, acc, false, false);
    }
    int col = col0 + (lane & 15);
    float bv = bias[col];
    int h = (lane >> 4) & 1;
#pragma unroll
    for (int v = 0; v < 8; ++v) {
        int row = row0 + v + 8 * h;
        Y[(size_t)row * CDIM + col] = acc[v] + bv;
    }
}

extern "C" void kernel_launch(void* const* d_in, const int* in_sizes, int n_in,
                              void* d_out, int out_size, void* d_ws, size_t ws_size,
                              hipStream_t stream) {
    const float* x  = (const float*)d_in[0];
    const float* Wf = (const float*)d_in[1];
    const float* bf = (const float*)d_in[2];
    const float* Wg = (const float*)d_in[3];
    const float* bg = (const float*)d_in[4];
    const float* Wh = (const float*)d_in[5];
    const float* bh = (const float*)d_in[6];
    const float* Wv = (const float*)d_in[7];
    const float* bv = (const float*)d_in[8];

    // workspace: fx, gx (row-major f16), hxT ([B][C][N] f16), o ([B][C][N] f16),
    // softadj f32.  Total ~33.7 MB — resident in MI455X's 192 MB L2.
    const size_t T = (size_t)BATCH * NTOK * CDIM;
    _Float16* fx  = (_Float16*)d_ws;
    _Float16* gx  = fx + T;
    _Float16* hxT = gx + T;
    _Float16* o   = hxT + T;
    float* softadj = (float*)(o + T);

    dim3 gProj(ROWS / 16, CDIM / 64), bProj(128);
    proj_x_kernel<0><<<gProj, bProj, 0, stream>>>(x, Wf, bf, fx);
    proj_x_kernel<0><<<gProj, bProj, 0, stream>>>(x, Wg, bg, gx);
    proj_x_kernel<1><<<gProj, bProj, 0, stream>>>(x, Wh, bh, hxT);

    row_stats_kernel<<<dim3(256), dim3(256), 0, stream>>>(fx, gx, softadj);
    attn_out_kernel<<<dim3(256), dim3(256), 0, stream>>>(fx, gx, hxT, softadj, o);

    proj_final_kernel<<<gProj, bProj, 0, stream>>>(o, Wv, bv, (float*)d_out);
}
